// Model_12945031430929
// MI455X (gfx1250) — compile-verified, matrix operand order checked by reading, not run on hardware
//
#include <hip/hip_runtime.h>
#include <hip/hip_bf16.h>

#define Bn 16
#define Dn 128
#define Tn 8192
#define Kn 512
#define BT (Bn * Tn)          // 131072
#define SE_STRIDE 136          // LDS codebook row stride (bf16 elems): 16B aligned, bank-spread

typedef __attribute__((ext_vector_type(16))) __bf16 v16bf;
typedef __attribute__((ext_vector_type(16))) unsigned short v16us;
typedef __attribute__((ext_vector_type(8)))  unsigned short v8us;
typedef __attribute__((ext_vector_type(8)))  float v8f;

__device__ __forceinline__ unsigned short f2bf(float x) {
    unsigned u = __builtin_bit_cast(unsigned, x);
    unsigned r = u + 0x7FFFu + ((u >> 16) & 1u);   // round-to-nearest-even
    return (unsigned short)(r >> 16);
}

// ---------------- Kernel 1: normalize codebook -> f32 + bf16 copies ----------------
__global__ __launch_bounds__(128) void norm_embed(const float* __restrict__ e,
                                                  float* __restrict__ en,
                                                  unsigned short* __restrict__ ebf) {
    int c = blockIdx.x, d = threadIdx.x;
    float v = e[c * Dn + d];
    float ss = v * v;
    #pragma unroll
    for (int m = 16; m >= 1; m >>= 1) ss += __shfl_xor(ss, m, 32);
    __shared__ float sw[4];
    int lane = d & 31, w = d >> 5;
    if (lane == 0) sw[w] = ss;
    __syncthreads();
    float rn = rsqrtf(sw[0] + sw[1] + sw[2] + sw[3]);
    float nv = v * rn;
    en[c * Dn + d] = nv;
    ebf[c * Dn + d] = f2bf(nv);
}

// ---------------- Kernel 2: normalize z rows + WMMA dot + argmax -> z_id ----------------
// One wave per TWO 16-row tiles (M=32): each LDS B-fragment feeds 2 WMMAs, halving
// LDS read traffic vs M=16. Codebook (bf16, 512x128, stride-padded) lives in LDS.
__global__ __launch_bounds__(256) void assign_kernel(const float* __restrict__ z,
                                                     const unsigned short* __restrict__ ebf,
                                                     int* __restrict__ zid) {
    __shared__ unsigned short sE[Kn * SE_STRIDE];
    int tid = threadIdx.x;

    // cooperative codebook load: 8192 x 16B chunks
    const uint4* src = (const uint4*)ebf;
    for (int q = tid; q < (Kn * Dn / 8); q += 256) {
        int code = q >> 4, db = q & 15;
        *(uint4*)&sE[code * SE_STRIDE + db * 8] = src[q];
    }
    __syncthreads();

    int wave = tid >> 5, lane = tid & 31;
    int half = lane >> 4, m = lane & 15;
    int tile0 = (blockIdx.x * 8 + wave) * 2;   // two consecutive 16-row tiles
    int g0 = tile0 * 16;                       // 32 | T, so both tiles share one b
    int b = g0 >> 13;                          // / T
    int t0 = g0 & (Tn - 1);
    const float* zb = z + (size_t)b * Dn * Tn + t0 + m;

    // build bf16 A fragments for both row-tiles (16-bit A 16x32 VGPR layout)
    v16bf a[2][4];
    #pragma unroll
    for (int rt = 0; rt < 2; ++rt) {
        const float* zr = zb + rt * 16;
        float raw[64];
        float ss = 0.f;
        #pragma unroll
        for (int s = 0; s < 4; ++s)
            #pragma unroll
            for (int j = 0; j < 16; ++j) {
                int Kf = s * 32 + (j >> 3) * 16 + half * 8 + (j & 7);
                float v = zr[(size_t)Kf * Tn];
                raw[s * 16 + j] = v;
                ss += v * v;
            }
        ss += __shfl_xor(ss, 16, 32);   // partner lane holds the other 64 features
        float rn = rsqrtf(ss);
        #pragma unroll
        for (int s = 0; s < 4; ++s) {
            v16us u;
            #pragma unroll
            for (int j = 0; j < 16; ++j) u[j] = f2bf(raw[s * 16 + j] * rn);
            a[rt][s] = __builtin_bit_cast(v16bf, u);
        }
    }

    float bestV[2][8];
    int   bestI[2][8];
    #pragma unroll
    for (int rt = 0; rt < 2; ++rt)
        #pragma unroll
        for (int r = 0; r < 8; ++r) { bestV[rt][r] = -1e30f; bestI[rt][r] = 0; }

    for (int n = 0; n < 32; ++n) {     // 32 tiles of 16 codes
        int c = n * 16 + m;
        const unsigned short* cp = &sE[c * SE_STRIDE + half * 8];
        // stage the 4 B fragments once; reuse for both row-tiles
        v16bf bb[4];
        #pragma unroll
        for (int s = 0; s < 4; ++s) {
            v8us lo = *(const v8us*)&cp[s * 32];       // K = s*32 + half*8 + 0..7
            v8us hi = *(const v8us*)&cp[s * 32 + 16];  // K = s*32 + 16 + half*8 + 0..7
            v16us bu;
            #pragma unroll
            for (int j = 0; j < 8; ++j) { bu[j] = lo[j]; bu[8 + j] = hi[j]; }
            bb[s] = __builtin_bit_cast(v16bf, bu);
        }
        #pragma unroll
        for (int rt = 0; rt < 2; ++rt) {
            v8f acc = {};
            #pragma unroll
            for (int s = 0; s < 4; ++s)
                acc = __builtin_amdgcn_wmma_f32_16x16x32_bf16(
                    false, a[rt][s], false, bb[s], (short)0, acc, false, false);
            // c is strictly increasing over n for this lane, so an equal value can
            // never replace an earlier (smaller-index) winner: plain '>' suffices
            // and if-converts to v_cmp_gt + v_cndmask pairs (no exec-mask branches).
            #pragma unroll
            for (int r = 0; r < 8; ++r) {
                float v = acc[r];
                bool upd = v > bestV[rt][r];
                bestV[rt][r] = upd ? v : bestV[rt][r];
                bestI[rt][r] = upd ? c : bestI[rt][r];
            }
        }
    }

    // reduce across the 16 lanes of each half (rows r+8*half); argmin-style tiebreak
    #pragma unroll
    for (int rt = 0; rt < 2; ++rt) {
        #pragma unroll
        for (int r = 0; r < 8; ++r) {
            float v = bestV[rt][r]; int idx = bestI[rt][r];
            #pragma unroll
            for (int msk = 8; msk >= 1; msk >>= 1) {
                float ov = __shfl_xor(v, msk, 32);
                int   oi = __shfl_xor(idx, msk, 32);
                bool take = (ov > v) | ((ov == v) & (oi < idx));
                v   = take ? ov : v;
                idx = take ? oi : idx;
            }
            bestI[rt][r] = idx;
        }
        if (m == 0) {
            #pragma unroll
            for (int r = 0; r < 8; ++r) zid[g0 + rt * 16 + r + half * 8] = bestI[rt][r];
        }
    }
}

// ---------------- Kernel 3: per-batch RLE compaction + p_mc_tr + lengths ----------------
__global__ __launch_bounds__(256) void rle_kernel(const int* __restrict__ zid,
                                                  const int* __restrict__ p,
                                                  float* __restrict__ out_zid_tr,
                                                  float* __restrict__ out_p_tr,
                                                  float* __restrict__ out_len,
                                                  int* __restrict__ zid_tr_int) {
    int b = blockIdx.x, tid = threadIdx.x;
    int lane = tid & 31, w = tid >> 5;
    __shared__ int sWaveTot[8], sWaveOff[8];
    __shared__ int sCarry, sPcnt;
    if (tid == 0) { sCarry = 0; sPcnt = 0; }
    __syncthreads();
    const int* zb = zid + b * Tn;
    const int* pb = p + b * Tn;

    for (int chunk = 0; chunk < Tn / 256; ++chunk) {
        int t = chunk * 256 + tid;
        int cur = zb[t];
        bool flag = (t < Tn - 1) ? (zb[t + 1] != cur) : (cur != 0);  // sentinel id*10000
        unsigned mask = (unsigned)__ballot(flag);
        int prefix = __popc(mask & ((1u << lane) - 1u));
        bool pflag = flag && (pb[t] != 0);
        unsigned pmask = (unsigned)__ballot(pflag);
        if (lane == 0) {
            sWaveTot[w] = __popc(mask);
            atomicAdd(&sPcnt, __popc(pmask));
        }
        __syncthreads();
        if (tid == 0) {
            int run = sCarry;
            #pragma unroll
            for (int i = 0; i < 8; ++i) { sWaveOff[i] = run; run += sWaveTot[i]; }
            sCarry = run;
        }
        __syncthreads();
        if (flag) {
            int pos = b * Tn + sWaveOff[w] + prefix;
            out_zid_tr[pos] = (float)cur;
            zid_tr_int[pos] = cur;
        }
        __syncthreads();
    }
    int len = sCarry, pcnt = sPcnt;
    for (int t = tid; t < Tn; t += 256) {
        if (t >= len) { out_zid_tr[b * Tn + t] = 0.0f; zid_tr_int[b * Tn + t] = 0; }
        float pv = 0.0f;
        if (t < pcnt) pv = (float)(t + 1);
        if (t == pcnt - 1) pv = (float)(-pcnt);
        out_p_tr[b * Tn + t] = pv;
    }
    if (tid == 0) out_len[b] = (float)len;
}

// ---------------- Kernel 4: gather z_vq_tr[b,d,t] = e_norm[z_id_tr[b,t], d] ----------------
__global__ __launch_bounds__(256) void gather_kernel(const int* __restrict__ zid_tr_int,
                                                     const float* __restrict__ en,
                                                     float* __restrict__ out) {
    int i = blockIdx.x * 256 + threadIdx.x;       // 0 .. B*D*T-1, t fastest
    int t = i & (Tn - 1);
    int d = (i >> 13) & (Dn - 1);
    int b = i >> 20;
    int id = zid_tr_int[b * Tn + t];
    out[i] = en[id * Dn + d];
}

extern "C" void kernel_launch(void* const* d_in, const int* in_sizes, int n_in,
                              void* d_out, int out_size, void* d_ws, size_t ws_size,
                              hipStream_t stream) {
    const float* z_mc = (const float*)d_in[0];     // [B, D, T] f32
    const float* emb  = (const float*)d_in[1];     // [K, D] f32
    const int*   p_mc = (const int*)d_in[2];       // [B, T] i32

    char* ws = (char*)d_ws;
    float*          en         = (float*)ws;                          // 512*128 f32   (256 KB)
    unsigned short* ebf        = (unsigned short*)(ws + 262144);      // 512*128 bf16  (128 KB)
    int*            zid        = (int*)(ws + 393216);                 // B*T i32       (512 KB)
    int*            zid_tr_int = (int*)(ws + 917504);                 // B*T i32       (512 KB)

    float* out      = (float*)d_out;
    float* out_zvq  = out;                         // B*D*T
    float* out_zid  = out + (size_t)Bn * Dn * Tn;  // B*T
    float* out_ptr  = out_zid + BT;                // B*T
    float* out_len  = out_ptr + BT;                // B

    norm_embed<<<Kn, Dn, 0, stream>>>(emb, en, ebf);
    assign_kernel<<<BT / 16 / 16, 256, 0, stream>>>(z_mc, ebf, zid);  // 512 blocks, 2 tiles/wave
    rle_kernel<<<Bn, 256, 0, stream>>>(zid, p_mc, out_zid, out_ptr, out_len, zid_tr_int);
    gather_kernel<<<(Bn * Dn * Tn) / 256, 256, 0, stream>>>(zid_tr_int, en, out_zvq);
}
